// GroupedQueryAttention_54202487275925
// MI455X (gfx1250) — compile-verified, matrix operand order checked by reading, not run on hardware
//
#include <hip/hip_runtime.h>
#include <hip/hip_bf16.h>
#include <stdint.h>

// ---------------------------------------------------------------------------
// GQA forward for MI455X (gfx1250): bf16 WMMA GEMMs + flash-style attention.
// B=2, T=2048, DIM=2048, H=16, KV=4, D=128, G=4
// - f32 -> bf16 via single v_perm_b32 per pair (keeps VALU off the critical
//   path so the WMMA pipe is the limiter)
// - GEMM uses LDS double-buffering + register prefetch: next tile's global
//   loads are in flight while the current tile's WMMAs execute; one barrier
//   per K-step.
// ---------------------------------------------------------------------------

typedef __attribute__((ext_vector_type(16))) __bf16 v16bf;
typedef __attribute__((ext_vector_type(8)))  float  v8f;

union Frag16 {
    uint4 q[2];
    v16bf v;
};

// one v_lshrrev_b32: truncate-to-bf16
__device__ __forceinline__ unsigned short f2bf(float x) {
    return (unsigned short)(__builtin_bit_cast(unsigned int, x) >> 16);
}
// one v_perm_b32: pack two floats' high halfwords -> packed bf16x2
__device__ __forceinline__ unsigned int pk2(float a, float b) {
    return __builtin_amdgcn_perm(__builtin_bit_cast(unsigned int, b),
                                 __builtin_bit_cast(unsigned int, a),
                                 0x07060302u);
}
__device__ __forceinline__ void ldfrag(Frag16& f, const unsigned short* p) {
    const uint4* q4 = (const uint4*)p;
    f.q[0] = q4[0];
    f.q[1] = q4[1];
}

// ---------------------------------------------------------------------------
// GEMM: C[M,N] = A[M,K] @ W[N,K]^T + bias[N]   (all f32 in memory, bf16 MACs)
// Block tile 128x128, K-step 32, 256 threads = 8 waves, wave tile 32x64.
// Double-buffered LDS; global loads of step s+1 issued before WMMAs of s.
// ---------------------------------------------------------------------------
__global__ __launch_bounds__(256) void gqa_gemm_bias(
    const float* __restrict__ A, const float* __restrict__ W,
    const float* __restrict__ bias, float* __restrict__ C,
    int M, int N, int K)
{
    __shared__ __align__(16) unsigned short As[2][128][32];
    __shared__ __align__(16) unsigned short Ws[2][128][32];

    const int tid  = threadIdx.x;
    const int lane = tid & 31;
    const int w    = tid >> 5;       // 0..7
    const int wm   = w & 3;          // wave row (32 rows each)
    const int wn   = w >> 2;         // wave col (64 cols each)
    const int rowBase = blockIdx.y * 128;
    const int colBase = blockIdx.x * 128;

    v8f acc[2][4];
#pragma unroll
    for (int i = 0; i < 2; ++i)
#pragma unroll
        for (int j = 0; j < 4; ++j)
#pragma unroll
            for (int e = 0; e < 8; ++e) acc[i][j][e] = 0.0f;

    const int lr = tid >> 1;         // 0..127 staging row
    const int lh = (tid & 1) * 16;   // 16-float half of the K-slab

    const int fr    = lane & 15;           // fragment row / output column
    const int khoff = (lane >> 4) * 16;    // K half owned by this lane

    const float* srcA0 = A + (size_t)(rowBase + lr) * K + lh;
    const float* srcW0 = W + (size_t)(colBase + lr) * K + lh;

    // --- prologue: stage K-step 0 into buffer 0 ---
    {
        float4 ra[4], rw[4];
#pragma unroll
        for (int m = 0; m < 4; ++m) {
            ra[m] = ((const float4*)srcA0)[m];
            rw[m] = ((const float4*)srcW0)[m];
        }
        unsigned int* dA = (unsigned int*)&As[0][lr][lh];
        unsigned int* dW = (unsigned int*)&Ws[0][lr][lh];
#pragma unroll
        for (int m = 0; m < 4; ++m) {
            dA[2 * m + 0] = pk2(ra[m].x, ra[m].y);
            dA[2 * m + 1] = pk2(ra[m].z, ra[m].w);
            dW[2 * m + 0] = pk2(rw[m].x, rw[m].y);
            dW[2 * m + 1] = pk2(rw[m].z, rw[m].w);
        }
    }
    __syncthreads();

    const int nsteps = K >> 5;
    int buf = 0;
    for (int s = 0; s < nsteps; ++s) {
        const bool more = (s + 1) < nsteps;

        // --- issue next tile's global loads first (latency hidden by WMMA) ---
        float4 ra[4], rw[4];
        if (more) {
            const float* srcA = srcA0 + (s + 1) * 32;
            const float* srcW = srcW0 + (s + 1) * 32;
#pragma unroll
            for (int m = 0; m < 4; ++m) {
                ra[m] = ((const float4*)srcA)[m];
                rw[m] = ((const float4*)srcW)[m];
            }
        }

        // --- fragments + WMMA from current buffer ---
        Frag16 af[2], bf[4];
#pragma unroll
        for (int i = 0; i < 2; ++i)
            ldfrag(af[i], &As[buf][wm * 32 + i * 16 + fr][khoff]);
#pragma unroll
        for (int j = 0; j < 4; ++j)
            ldfrag(bf[j], &Ws[buf][wn * 64 + j * 16 + fr][khoff]);

#pragma unroll
        for (int i = 0; i < 2; ++i)
#pragma unroll
            for (int j = 0; j < 4; ++j)
                acc[i][j] = __builtin_amdgcn_wmma_f32_16x16x32_bf16(
                    false, af[i].v, false, bf[j].v,
                    (short)0, acc[i][j], false, false);

        // --- convert + store next tile into the other buffer ---
        if (more) {
            unsigned int* dA = (unsigned int*)&As[buf ^ 1][lr][lh];
            unsigned int* dW = (unsigned int*)&Ws[buf ^ 1][lr][lh];
#pragma unroll
            for (int m = 0; m < 4; ++m) {
                dA[2 * m + 0] = pk2(ra[m].x, ra[m].y);
                dA[2 * m + 1] = pk2(ra[m].z, ra[m].w);
                dW[2 * m + 0] = pk2(rw[m].x, rw[m].y);
                dW[2 * m + 1] = pk2(rw[m].z, rw[m].w);
            }
        }
        __syncthreads();
        buf ^= 1;
    }

    // --- epilogue: C-layout (lane&15 = N col, (lane>>4)*8+ii = M row) ---
    const int nn = lane & 15;
    const int mb = (lane >> 4) * 8;
#pragma unroll
    for (int j = 0; j < 4; ++j) {
        int col = colBase + wn * 64 + j * 16 + nn;
        float bv = bias[col];
#pragma unroll
        for (int i = 0; i < 2; ++i) {
            int row = rowBase + wm * 32 + i * 16 + mb;
#pragma unroll
            for (int ii = 0; ii < 8; ++ii)
                C[(size_t)(row + ii) * N + col] = acc[i][j][ii] + bv;
        }
    }
}

// ---------------------------------------------------------------------------
// RoPE. Reference quirk: cos/sin are sliced by HEAD index on the position
// axis, so position = h for q, position = kv for k.
// ---------------------------------------------------------------------------
__global__ __launch_bounds__(256) void gqa_rope(float* __restrict__ q,
                                                float* __restrict__ k)
{
    const long long QP = 4096LL * 16 * 64;
    const long long KP = 4096LL * 4 * 64;
    long long idx = (long long)blockIdx.x * blockDim.x + threadIdx.x;
    // inv_freq[j] = 10000^(-j/64) = exp(-j * ln(10000)/64)
    const float nlf = -0.14391156463f;
    if (idx < QP) {
        int j   = (int)(idx & 63);
        int h   = (int)((idx >> 6) & 15);
        long long row = idx >> 10;
        float* base = q + row * 2048 + h * 128;
        float ang = (float)h * __expf(nlf * (float)j);
        float s, c; __sincosf(ang, &s, &c);
        float x1 = base[j], x2 = base[j + 64];
        base[j]      = x1 * c - x2 * s;
        base[j + 64] = x2 * c + x1 * s;
    } else if (idx < QP + KP) {
        long long i2 = idx - QP;
        int j    = (int)(i2 & 63);
        int kvh  = (int)((i2 >> 6) & 3);
        long long row = i2 >> 8;
        float* base = k + row * 512 + kvh * 128;
        float ang = (float)kvh * __expf(nlf * (float)j);
        float s, c; __sincosf(ang, &s, &c);
        float x1 = base[j], x2 = base[j + 64];
        base[j]      = x1 * c - x2 * s;
        base[j + 64] = x2 * c + x1 * s;
    }
}

// ---------------------------------------------------------------------------
// Flash-style attention. One workgroup = 128 query rows of one (b,h).
// 8 waves x 16 rows. Streams 64-key blocks; online softmax; note the
// reference mask ADDS 1.0 on the lower triangle (no -inf masking), so we
// sweep all 2048 keys.
// ---------------------------------------------------------------------------
__global__ __launch_bounds__(256) void gqa_attn(
    const float* __restrict__ q, const float* __restrict__ k,
    const float* __restrict__ v, float* __restrict__ o)
{
    __shared__ __align__(16) unsigned short Ks[64][128];     // [s][d]
    __shared__ __align__(16) unsigned short VsT[128][64];    // [d][s] transposed
    __shared__ __align__(16) unsigned short Ps[8][16][64];   // per-wave P scratch

    const int tid  = threadIdx.x;
    const int lane = tid & 31;
    const int w    = tid >> 5;
    const int b    = blockIdx.y >> 4;
    const int h    = blockIdx.y & 15;
    const int kvh  = h >> 2;
    const int t0   = blockIdx.x * 128;
    const float scale = 0.088388347648f;   // 128^-0.5

    const int fr    = lane & 15;
    const int khoff = (lane >> 4) * 16;

    // --- Q fragments for this wave's 16 rows, held in registers as bf16 ---
    Frag16 qf[4];
    {
        const float* qrow =
            q + (size_t)(b * 2048 + t0 + w * 16 + fr) * 2048 + h * 128;
#pragma unroll
        for (int s = 0; s < 4; ++s) {
            const float4* p = (const float4*)(qrow + s * 32 + khoff);
            float4 f0 = p[0], f1 = p[1], f2 = p[2], f3 = p[3];
            qf[s].q[0] = make_uint4(pk2(f0.x, f0.y), pk2(f0.z, f0.w),
                                    pk2(f1.x, f1.y), pk2(f1.z, f1.w));
            qf[s].q[1] = make_uint4(pk2(f2.x, f2.y), pk2(f2.z, f2.w),
                                    pk2(f3.x, f3.y), pk2(f3.z, f3.w));
        }
    }

    v8f accO[8];
#pragma unroll
    for (int dt = 0; dt < 8; ++dt)
#pragma unroll
        for (int e = 0; e < 8; ++e) accO[dt][e] = 0.0f;

    float mrow[8], lrow[8];
#pragma unroll
    for (int ii = 0; ii < 8; ++ii) { mrow[ii] = -3.0e38f; lrow[ii] = 0.0f; }

    const int nn = lane & 15;
    const int mb = (lane >> 4) * 8;

    for (int s0 = 0; s0 < 2048; s0 += 64) {
        // --- cooperative staging of K (d-contiguous) and V (transposed) ---
        {
            const int r = tid >> 2;
            const int c = (tid & 3) * 32;
            const float* srck =
                k + (size_t)(b * 2048 + s0 + r) * 512 + kvh * 128 + c;
            unsigned int* dK = (unsigned int*)&Ks[r][c];
#pragma unroll
            for (int m = 0; m < 8; ++m) {
                float4 f = ((const float4*)srck)[m];
                dK[2 * m + 0] = pk2(f.x, f.y);
                dK[2 * m + 1] = pk2(f.z, f.w);
            }
            const float* srcv =
                v + (size_t)(b * 2048 + s0 + r) * 512 + kvh * 128 + c;
#pragma unroll
            for (int m = 0; m < 8; ++m) {
                float4 f = ((const float4*)srcv)[m];
                VsT[c + 4 * m + 0][r] = f2bf(f.x);
                VsT[c + 4 * m + 1][r] = f2bf(f.y);
                VsT[c + 4 * m + 2][r] = f2bf(f.z);
                VsT[c + 4 * m + 3][r] = f2bf(f.w);
            }
            if (s0 + 64 < 2048) {
                __builtin_prefetch(srck + 64 * 512, 0, 0); // global_prefetch_b8
                __builtin_prefetch(srcv + 64 * 512, 0, 0);
            }
        }
        __syncthreads();

        // --- S = Q @ K^T over this 64-key block (4 column tiles) ---
        v8f S[4];
#pragma unroll
        for (int st = 0; st < 4; ++st) {
#pragma unroll
            for (int e = 0; e < 8; ++e) S[st][e] = 0.0f;
#pragma unroll
            for (int ks = 0; ks < 4; ++ks) {
                Frag16 kf;
                ldfrag(kf, &Ks[st * 16 + fr][ks * 32 + khoff]);
                S[st] = __builtin_amdgcn_wmma_f32_16x16x32_bf16(
                    false, qf[ks].v, false, kf.v,
                    (short)0, S[st], false, false);
            }
        }

        // --- scale + (+1 lower-triangle) mask, block row max ---
        float bmax[8];
#pragma unroll
        for (int ii = 0; ii < 8; ++ii) bmax[ii] = -3.0e38f;
#pragma unroll
        for (int st = 0; st < 4; ++st) {
            int scol = s0 + st * 16 + nn;
#pragma unroll
            for (int ii = 0; ii < 8; ++ii) {
                int trow = t0 + w * 16 + mb + ii;
                float sv = S[st][ii] * scale + ((scol <= trow) ? 1.0f : 0.0f);
                S[st][ii] = sv;
                bmax[ii] = fmaxf(bmax[ii], sv);
            }
        }
#pragma unroll
        for (int off = 1; off < 16; off <<= 1)
#pragma unroll
            for (int ii = 0; ii < 8; ++ii)
                bmax[ii] = fmaxf(bmax[ii], __shfl_xor(bmax[ii], off, 32));

        // --- online softmax rescale ---
        float fac[8];
#pragma unroll
        for (int ii = 0; ii < 8; ++ii) {
            float mn = fmaxf(mrow[ii], bmax[ii]);
            fac[ii] = __expf(mrow[ii] - mn);
            mrow[ii] = mn;
            lrow[ii] *= fac[ii];
        }
#pragma unroll
        for (int dt = 0; dt < 8; ++dt)
#pragma unroll
            for (int ii = 0; ii < 8; ++ii) accO[dt][ii] *= fac[ii];

        // --- P = exp(S - m), row sums, spill P to per-wave LDS as bf16 ---
        float rsum[8];
#pragma unroll
        for (int ii = 0; ii < 8; ++ii) rsum[ii] = 0.0f;
#pragma unroll
        for (int st = 0; st < 4; ++st)
#pragma unroll
            for (int ii = 0; ii < 8; ++ii) {
                float p = __expf(S[st][ii] - mrow[ii]);
                rsum[ii] += p;
                Ps[w][mb + ii][st * 16 + nn] = f2bf(p);
            }
#pragma unroll
        for (int off = 1; off < 16; off <<= 1)
#pragma unroll
            for (int ii = 0; ii < 8; ++ii)
                rsum[ii] += __shfl_xor(rsum[ii], off, 32);
#pragma unroll
        for (int ii = 0; ii < 8; ++ii) lrow[ii] += rsum[ii];

        // --- O += P @ V (V already transposed in LDS) ---
#pragma unroll
        for (int ks2 = 0; ks2 < 2; ++ks2) {
            Frag16 pf;
            ldfrag(pf, &Ps[w][fr][ks2 * 32 + khoff]);
#pragma unroll
            for (int dt = 0; dt < 8; ++dt) {
                Frag16 vf;
                ldfrag(vf, &VsT[dt * 16 + fr][ks2 * 32 + khoff]);
                accO[dt] = __builtin_amdgcn_wmma_f32_16x16x32_bf16(
                    false, pf.v, false, vf.v,
                    (short)0, accO[dt], false, false);
            }
        }
        __syncthreads();
    }

    // --- normalize and store ---
    float inv[8];
#pragma unroll
    for (int ii = 0; ii < 8; ++ii) inv[ii] = 1.0f / lrow[ii];
#pragma unroll
    for (int dt = 0; dt < 8; ++dt)
#pragma unroll
        for (int ii = 0; ii < 8; ++ii) {
            size_t idx = (size_t)(b * 2048 + t0 + w * 16 + mb + ii) * 2048 +
                         h * 128 + dt * 16 + nn;
            o[idx] = accO[dt][ii] * inv[ii];
        }
}

// ---------------------------------------------------------------------------
extern "C" void kernel_launch(void* const* d_in, const int* in_sizes, int n_in,
                              void* d_out, int out_size, void* d_ws, size_t ws_size,
                              hipStream_t stream) {
    const float* x  = (const float*)d_in[0];
    const float* wq = (const float*)d_in[1];
    const float* bq = (const float*)d_in[2];
    const float* wk = (const float*)d_in[3];
    const float* bk = (const float*)d_in[4];
    const float* wv = (const float*)d_in[5];
    const float* bv = (const float*)d_in[6];
    const float* wo = (const float*)d_in[7];
    const float* bo = (const float*)d_in[8];
    float* out = (float*)d_out;

    const int BT = 4096;   // B*T
    float* qb = (float*)d_ws;                     // 4096 x 2048
    float* kb = qb + (size_t)BT * 2048;           // 4096 x 512
    float* vb = kb + (size_t)BT * 512;            // 4096 x 512
    float* ab = vb + (size_t)BT * 512;            // 4096 x 2048

    // QKV projections (bias fused)
    gqa_gemm_bias<<<dim3(2048 / 128, BT / 128), 256, 0, stream>>>(
        x, wq, bq, qb, BT, 2048, 2048);
    gqa_gemm_bias<<<dim3(512 / 128, BT / 128), 256, 0, stream>>>(
        x, wk, bk, kb, BT, 512, 2048);
    gqa_gemm_bias<<<dim3(512 / 128, BT / 128), 256, 0, stream>>>(
        x, wv, bv, vb, BT, 512, 2048);

    // RoPE on q and k (in place)
    {
        long long total = 4096LL * 16 * 64 + 4096LL * 4 * 64;
        int blocks = (int)((total + 255) / 256);
        gqa_rope<<<blocks, 256, 0, stream>>>(qb, kb);
    }

    // Attention: grid (T/128, B*H)
    gqa_attn<<<dim3(16, 32), 256, 0, stream>>>(qb, kb, vb, ab);

    // Output projection
    gqa_gemm_bias<<<dim3(2048 / 128, BT / 128), 256, 0, stream>>>(
        ab, wo, bo, out, BT, 2048, 2048);
}
